// DotProdAttention_90924457656650
// MI455X (gfx1250) — matrix operand order repaired and checked by reading, not compiled, not run on hardware
//
#include <hip/hip_runtime.h>

typedef __attribute__((ext_vector_type(16))) _Float16 v16h;
typedef __attribute__((ext_vector_type(8)))  _Float16 v8h;
typedef __attribute__((ext_vector_type(2)))  _Float16 v2h;
typedef __attribute__((ext_vector_type(8)))  float    v8f;

#define BATCH 32
#define SEQ   2048
#define DHEAD 64

// 1/sqrt(64) * log2(e): softmax computed in base-2 domain (v_exp_f32 is exp2).
#define QSCALE 0.18033688011112042f

// ---------------------------------------------------------------------------
// Pre-pass: K (f32, [b][s][d]) -> Kh (f16, [b][s][d])
//           V (f32, [b][s][d]) -> Vt (f16, [b][d][s])  (transposed)
// ---------------------------------------------------------------------------
__global__ __launch_bounds__(256) void cvt_kv(const float* __restrict__ K,
                                              const float* __restrict__ V,
                                              _Float16* __restrict__ Kh,
                                              _Float16* __restrict__ Vt) {
    __shared__ float tile[64][65];
    const int b  = blockIdx.y;
    const int s0 = blockIdx.x * 64;
    const int t  = threadIdx.x;
    const int dc = t & 63;
    const int rq = t >> 6;

    const float* Kb  = K  + ((size_t)b * SEQ + s0) * DHEAD;
    const float* Vb  = V  + ((size_t)b * SEQ + s0) * DHEAD;
    _Float16*    Khb = Kh + ((size_t)b * SEQ + s0) * DHEAD;

    #pragma unroll
    for (int it = 0; it < 16; ++it) {
        const int r = it * 4 + rq;
        Khb[(size_t)r * DHEAD + dc] = (_Float16)Kb[(size_t)r * DHEAD + dc];
        tile[r][dc] = Vb[(size_t)r * DHEAD + dc];
    }
    __syncthreads();

    _Float16* Vtb = Vt + (size_t)b * SEQ * DHEAD;
    #pragma unroll
    for (int it = 0; it < 16; ++it) {
        const int d = it * 4 + rq;
        Vtb[(size_t)d * SEQ + s0 + dc] = (_Float16)tile[dc][d];
    }
}

__device__ __forceinline__ unsigned pkh(float a, float b) {
    v2h t; t[0] = (_Float16)a; t[1] = (_Float16)b;
    return __builtin_bit_cast(unsigned, t);
}
__device__ __forceinline__ v16h cat8(v8h lo, v8h hi) {
    return __builtin_shufflevector(lo, hi, 0,1,2,3,4,5,6,7,8,9,10,11,12,13,14,15);
}
__device__ __forceinline__ float rmax8(v8f x) {
    return fmaxf(fmaxf(fmaxf(x[0], x[1]), fmaxf(x[2], x[3])),
                 fmaxf(fmaxf(x[4], x[5]), fmaxf(x[6], x[7])));
}
__device__ __forceinline__ float rsum8(v8f x) {
    return ((x[0] + x[1]) + (x[2] + x[3])) + ((x[4] + x[5]) + (x[6] + x[7]));
}

union U16 { v16h h; unsigned u[8]; };

// ---------------------------------------------------------------------------
// Transposed flash attention: S^T = K @ Q^T, O^T = V^T @ P^T.
// Each query lives in one lane column -> softmax reductions are in-lane
// trees + a single xor-16 shuffle; P^T (C-layout) -> B-layout needs only a
// half-wave f16x2 exchange. No LDS in the hot loop.
// Block = 128 threads = 4 waves; wave owns 16 queries; Bc = 64 keys/iter.
// ---------------------------------------------------------------------------
__global__ __launch_bounds__(128) void fattn_f16_wmma(const float* __restrict__ Q,
                                                      const _Float16* __restrict__ Kh,
                                                      const _Float16* __restrict__ Vt,
                                                      float* __restrict__ Out) {
    __shared__ float ldsO[4][16][68];      // epilogue transpose only (padded)

    const int lane = threadIdx.x & 31;
    const int w    = threadIdx.x >> 5;
    const int grp  = lane >> 4;
    const int l16  = lane & 15;

    const int b  = blockIdx.y;
    const int q0 = blockIdx.x * 64 + w * 16;   // wave's 16 queries: q0 + l16

    const float*    Qb  = Q  + (size_t)b * SEQ * DHEAD;
    const _Float16* Khb = Kh + (size_t)b * SEQ * DHEAD;
    const _Float16* Vtb = Vt + (size_t)b * SEQ * DHEAD;
    float*          Ob  = Out + (size_t)b * SEQ * DHEAD;

    // Q^T fragments in B layout (col N = query = l16, K = dim = 32c+16*grp+h),
    // contiguous in row-major Q; pre-scaled by 1/sqrt(d) * log2(e).
    v16h bq[2];
    {
        const float* qrow = Qb + (size_t)(q0 + l16) * DHEAD;
        #pragma unroll
        for (int c = 0; c < 2; ++c) {
            const int d0 = 32 * c + 16 * grp;
            #pragma unroll
            for (int h = 0; h < 16; ++h)
                bq[c][h] = (_Float16)(qrow[d0 + h] * QSCALE);
        }
    }

    v8f o[4] = {};            // O^T: 4 C-tiles, row = dim 16j+r+8*grp, col = query l16
    float m = -1e30f, l = 0.0f;   // per-lane (per-query) running max / sum (base-2)

    for (int kt = 0; kt < SEQ; kt += 64) {
        // ---- K fragments (A layout): row M = key, two 16B runs per chunk ----
        v16h ak[4][2];
        #pragma unroll
        for (int n = 0; n < 4; ++n) {
            const _Float16* kp = Khb + (size_t)(kt + 16 * n + l16) * DHEAD + 16 * grp;
            #pragma unroll
            for (int c = 0; c < 2; ++c)
                ak[n][c] = cat8(*(const v8h*)(kp + 32 * c),
                                *(const v8h*)(kp + 32 * c + 8));
        }
        // prefetch 2 tiles ahead (the kt+64 loads already pipeline normally)
        if (kt + 128 < SEQ) {
            __builtin_prefetch(Khb + (size_t)(kt + 128 + lane * 2) * DHEAD, 0, 1);
            __builtin_prefetch(Vtb + (size_t)(lane * 2) * SEQ + kt + 128, 0, 1);
        }

        // ---- S^T tiles: row = key 16n+r+8*grp, col = query l16 ----
        v8f s[4];
        #pragma unroll
        for (int n = 0; n < 4; ++n) {
            v8f acc = {};
            acc = __builtin_amdgcn_wmma_f32_16x16x32_f16(false, ak[n][0], false, bq[0],
                                                         (short)0, acc, false, false);
            acc = __builtin_amdgcn_wmma_f32_16x16x32_f16(false, ak[n][1], false, bq[1],
                                                         (short)0, acc, false, false);
            s[n] = acc;
        }

        // ---- V^T fragments (A layout) issued now; drain under softmax ----
        v16h av[4][2];
        #pragma unroll
        for (int j = 0; j < 4; ++j) {
            const _Float16* vp = Vtb + (size_t)(16 * j + l16) * SEQ + kt + 16 * grp;
            #pragma unroll
            for (int c = 0; c < 2; ++c)
                av[j][c] = cat8(*(const v8h*)(vp + 32 * c),
                                *(const v8h*)(vp + 32 * c + 8));
        }

        // ---- online softmax (base-2), per-lane scalars; one xor-16 each ----
        float v = fmaxf(fmaxf(rmax8(s[0]), rmax8(s[1])),
                        fmaxf(rmax8(s[2]), rmax8(s[3])));
        v = fmaxf(v, __shfl_xor(v, 16, 32));
        const float mnew  = fmaxf(m, v);
        const float alpha = __builtin_amdgcn_exp2f(m - mnew);
        #pragma unroll
        for (int n = 0; n < 4; ++n)
            #pragma unroll
            for (int r = 0; r < 8; ++r)
                s[n][r] = __builtin_amdgcn_exp2f(s[n][r] - mnew);
        float rs = (rsum8(s[0]) + rsum8(s[1])) + (rsum8(s[2]) + rsum8(s[3]));
        rs += __shfl_xor(rs, 16, 32);
        l = l * alpha + rs;
        m = mnew;

        // ---- P^T: C-layout -> B-layout via half-wave f16x2 exchange ----
        // Lane (grp,l16) B-frag chunk c: half h -> key kt+32c+16*grp+h,
        // i.e. tile n = 2c+grp; h<8 sourced from group-0 lanes, h>=8 from group-1.
        v16h bp[2];
        #pragma unroll
        for (int c = 0; c < 2; ++c) {
            unsigned p0[4], p1[4];
            #pragma unroll
            for (int d = 0; d < 4; ++d) {
                p0[d] = pkh(s[2 * c][2 * d],     s[2 * c][2 * d + 1]);
                p1[d] = pkh(s[2 * c + 1][2 * d], s[2 * c + 1][2 * d + 1]);
            }
            U16 fb;
            #pragma unroll
            for (int d = 0; d < 4; ++d) {
                const unsigned send = grp ? p0[d] : p1[d];   // tile the partner needs
                const unsigned recv = (unsigned)__shfl_xor((int)send, 16, 32);
                fb.u[d]     = grp ? recv  : p0[d];           // h<8  (group-0 rows)
                fb.u[d + 4] = grp ? p1[d] : recv;            // h>=8 (group-1 rows)
            }
            bp[c] = fb.h;
        }

        // ---- O^T = O^T*alpha + V^T @ P^T ----
        #pragma unroll
        for (int j = 0; j < 4; ++j) {
            #pragma unroll
            for (int r = 0; r < 8; ++r) o[j][r] *= alpha;
            o[j] = __builtin_amdgcn_wmma_f32_16x16x32_f16(false, av[j][0], false, bp[0],
                                                          (short)0, o[j], false, false);
            o[j] = __builtin_amdgcn_wmma_f32_16x16x32_f16(false, av[j][1], false, bp[1],
                                                          (short)0, o[j], false, false);
        }
    }

    // ---- epilogue: normalize, transpose O^T via per-wave LDS, coalesced store ----
    const float inv = 1.0f / l;
    #pragma unroll
    for (int j = 0; j < 4; ++j) {
        #pragma unroll
        for (int r = 0; r < 8; ++r)
            ldsO[w][l16][16 * j + r + 8 * grp] = o[j][r] * inv;   // row = query
    }
    asm volatile("s_wait_dscnt 0" ::: "memory");
    #pragma unroll
    for (int it = 0; it < 8; ++it) {
        const int flat = it * 128 + lane * 4;
        const int row  = flat >> 6;          // query row within wave tile
        const int col  = flat & 63;          // dim
        float4 val = *(const float4*)&ldsO[w][row][col];
        *(float4*)&Ob[(size_t)(q0 + row) * DHEAD + col] = val;
    }
}

extern "C" void kernel_launch(void* const* d_in, const int* in_sizes, int n_in,
                              void* d_out, int out_size, void* d_ws, size_t ws_size,
                              hipStream_t stream) {
    (void)in_sizes; (void)n_in; (void)out_size; (void)ws_size;
    const float* Q = (const float*)d_in[0];
    const float* K = (const float*)d_in[1];
    const float* V = (const float*)d_in[2];
    float* O = (float*)d_out;

    _Float16* Kh = (_Float16*)d_ws;                                 // 8 MB
    _Float16* Vt = Kh + (size_t)BATCH * SEQ * DHEAD;                // 8 MB

    dim3 grid(SEQ / 64, BATCH);   // (32, 32)
    cvt_kv<<<grid, 256, 0, stream>>>(K, V, Kh, Vt);
    fattn_f16_wmma<<<grid, 128, 0, stream>>>(Q, Kh, Vt, O);
}
